// TurboGCN_8881992368458
// MI455X (gfx1250) — matrix operand order repaired and checked by reading, not compiled, not run on hardware
//
#include <hip/hip_runtime.h>
#include <stdint.h>

typedef __attribute__((ext_vector_type(16))) __bf16 v16bf;
typedef __attribute__((ext_vector_type(8)))  float  v8f;

#define HDIM 256

__device__ __forceinline__ unsigned short f2bf(float f) {
  unsigned int u = __float_as_uint(f);
  u += 0x7FFFu + ((u >> 16) & 1u);            // round-to-nearest-even
  return (unsigned short)(u >> 16);
}
__device__ __forceinline__ float bf2f(unsigned short h) {
  return __uint_as_float(((unsigned int)h) << 16);
}
__device__ __forceinline__ float elu(float z) {
  return z > 0.f ? z : (__expf(z) - 1.f);
}
__device__ __forceinline__ void atomAddF(float* p, float v) {
  __hip_atomic_fetch_add(p, v, __ATOMIC_RELAXED, __HIP_MEMORY_SCOPE_AGENT);
}

union ABPack { uint4 u[2]; v16bf v; };

// ---------------- degree / dinv ----------------
__global__ void deg_count(const int* __restrict__ dst, unsigned int* __restrict__ deg, int E) {
  int e = blockIdx.x * blockDim.x + threadIdx.x;
  if (e < E)
    __hip_atomic_fetch_add(&deg[dst[e]], 1u, __ATOMIC_RELAXED, __HIP_MEMORY_SCOPE_AGENT);
}
__global__ void deg_to_dinv(float* __restrict__ buf, int N) {
  int i = blockIdx.x * blockDim.x + threadIdx.x;
  if (i < N) {
    unsigned int u = ((unsigned int*)buf)[i];
    buf[i] = rsqrtf((float)u + 1.0f);        // deg includes self-loop (+1)
  }
}

// ---------------- weight pack: W[k][n] (f32) -> Wt[n][k] (bf16) ----------------
__global__ void pack_wt(const float* __restrict__ W, unsigned short* __restrict__ wt) {
  int idx = blockIdx.x * blockDim.x + threadIdx.x;   // idx = k*H + n
  int k = idx >> 8, n = idx & (HDIM - 1);
  wt[n * HDIM + k] = f2bf(W[idx]);
}

// ---------------- input proj + LayerNorm + ELU -> bf16 ----------------
__global__ void input_proj(const float* __restrict__ ctrl, const float* __restrict__ w_in,
                           const float* __restrict__ b_in, const float* __restrict__ g,
                           const float* __restrict__ be, unsigned short* __restrict__ xb) {
  __shared__ float red[HDIM];
  int n = blockIdx.x, c = threadIdx.x;
  float v = ctrl[n] * w_in[c] + b_in[c];
  red[c] = v;
  __syncthreads();
  for (int s = HDIM / 2; s > 0; s >>= 1) { if (c < s) red[c] += red[c + s]; __syncthreads(); }
  float mu = red[0] * (1.0f / HDIM);
  __syncthreads();
  float d = v - mu;
  red[c] = d * d;
  __syncthreads();
  for (int s = HDIM / 2; s > 0; s >>= 1) { if (c < s) red[c] += red[c + s]; __syncthreads(); }
  float var = red[0] * (1.0f / HDIM);
  float y = d * rsqrtf(var + 1e-5f) * g[c] + be[c];
  xb[(size_t)n * HDIM + c] = f2bf(elu(y));
}

// ---------------- GEMM: xw[N,H] = xb[N,H](bf16) @ W  via v_wmma_f32_16x16x32_bf16 --------
// block = 128 threads (4 waves); wave w computes cols [w*64, w*64+64) of one 16-row stripe.
__global__ void gemm_xw(const unsigned short* __restrict__ xb,
                        const unsigned short* __restrict__ wt,   // Wt[n][k], bf16
                        float* __restrict__ xw, int N) {
  const int lane = threadIdx.x & 31;
  const int wave = threadIdx.x >> 5;
  const int m    = lane & 15;
  const int half = lane >> 4;
  const int row0 = blockIdx.x << 4;
  int rl = row0 + m;
  if (rl >= N) rl = N - 1;                    // branchless clamp; keeps EXEC all-ones
  const unsigned short* arow = xb + (size_t)rl * HDIM;
  const int n0 = wave * 64;

  v8f acc[4] = {};
  for (int k0 = 0; k0 < HDIM; k0 += 32) {
    // A tile 16x32: lane holds row m, K = [8h,8h+8) and [16+8h,16+8h+8)
    ABPack a;
    a.u[0] = *(const uint4*)(arow + k0 + 8 * half);
    a.u[1] = *(const uint4*)(arow + k0 + 16 + 8 * half);
    const unsigned short* bp = wt + (size_t)(n0 + m) * HDIM + k0 + 8 * half;
#pragma unroll
    for (int t = 0; t < 4; ++t) {
      // B tile 32x16: lane holds col n0+t*16+m, K = [8h,8h+8) and [16+8h,16+8h+8)
      ABPack b;
      b.u[0] = *(const uint4*)(bp + (size_t)t * 16 * HDIM);
      b.u[1] = *(const uint4*)(bp + (size_t)t * 16 * HDIM + 16);
      acc[t] = __builtin_amdgcn_wmma_f32_16x16x32_bf16(false, a.v, false, b.v,
                                                       (short)0, acc[t], false, false);
    }
  }
#pragma unroll
  for (int r = 0; r < 8; ++r) {
    int row = row0 + r + 8 * half;            // C/D layout: VGPR r -> M = r + 8*half
    if (row < N) {
      float* orow = xw + (size_t)row * HDIM + n0 + m;
      orow[0]  = acc[0][r];
      orow[16] = acc[1][r];
      orow[32] = acc[2][r];
      orow[48] = acc[3][r];
    }
  }
}

// ---------------- agg = xw * dinv^2 + bias (self-loop + bias init) ----------------
__global__ void init_agg(const float* __restrict__ xw, const float* __restrict__ dinv,
                         const float* __restrict__ b, float* __restrict__ agg) {
  int n = blockIdx.x, c = threadIdx.x;
  size_t i = (size_t)n * HDIM + c;
  float di = dinv[n];
  agg[i] = xw[i] * di * di + b[c];
}

// ---------------- edge scatter: agg[dst] += xw[src] * dinv[src]*dinv[dst] ----------------
__global__ void scatter(const int* __restrict__ src, const int* __restrict__ dst,
                        const float* __restrict__ dinv, const float* __restrict__ xw,
                        float* __restrict__ agg, int E) {
  int gid = blockIdx.x * blockDim.x + threadIdx.x;
  int e = gid >> 5;                            // one wave32 per edge
  int lane = gid & 31;
  if (e >= E) return;
  int s = src[e], d = dst[e];
  float coef = dinv[s] * dinv[d];
  const float4* xs = (const float4*)(xw + (size_t)s * HDIM);
  float* ag = agg + (size_t)d * HDIM;
#pragma unroll
  for (int i = 0; i < 2; ++i) {
    int q = lane + 32 * i;
    float4 v = xs[q];
    atomAddF(ag + 4 * q + 0, v.x * coef);
    atomAddF(ag + 4 * q + 1, v.y * coef);
    atomAddF(ag + 4 * q + 2, v.z * coef);
    atomAddF(ag + 4 * q + 3, v.w * coef);
  }
}

// ---------------- BN stats: per-channel sum & sumsq ----------------
__global__ void bn_stats(const float* __restrict__ agg, float* __restrict__ sums, int N) {
  int c = threadIdx.x;
  int r0 = blockIdx.x * 64;
  int r1 = r0 + 64; if (r1 > N) r1 = N;
  float s = 0.f, s2 = 0.f;
  for (int r = r0; r < r1; ++r) {
    float v = agg[(size_t)r * HDIM + c];
    s += v; s2 += v * v;
  }
  atomAddF(&sums[c], s);
  atomAddF(&sums[HDIM + c], s2);
}
__global__ void bn_coef(const float* __restrict__ sums, const float* __restrict__ g,
                        const float* __restrict__ be, float* __restrict__ coef, int N) {
  int c = threadIdx.x;
  float inv = 1.0f / (float)N;
  float mean = sums[c] * inv;
  float var  = sums[HDIM + c] * inv - mean * mean;
  float a = g[c] * rsqrtf(var + 1e-5f);
  coef[c] = a;
  coef[HDIM + c] = be[c] - mean * a;
}
__global__ void bn_apply(const float* __restrict__ agg, const float* __restrict__ coef,
                         unsigned short* __restrict__ xb) {
  int n = blockIdx.x, c = threadIdx.x;
  size_t i = (size_t)n * HDIM + c;
  float y = agg[i] * coef[c] + coef[HDIM + c];
  xb[i] = f2bf(elu(y));
}

// ---------------- head: out[n] = dot(x[n,:], w_head) + b_head ----------------
__global__ void head_k(const unsigned short* __restrict__ xb, const float* __restrict__ w,
                       const float* __restrict__ bh, float* __restrict__ out) {
  __shared__ float red[HDIM];
  int n = blockIdx.x, c = threadIdx.x;
  red[c] = bf2f(xb[(size_t)n * HDIM + c]) * w[c];
  __syncthreads();
  for (int s = HDIM / 2; s > 0; s >>= 1) { if (c < s) red[c] += red[c + s]; __syncthreads(); }
  if (c == 0) out[n] = red[0] + bh[0];
}

extern "C" void kernel_launch(void* const* d_in, const int* in_sizes, int n_in,
                              void* d_out, int out_size, void* d_ws, size_t ws_size,
                              hipStream_t stream) {
  const int N = in_sizes[0];
  const int E = in_sizes[1] / 2;

  const float* ctrl = (const float*)d_in[0];
  const int*   edge = (const int*)d_in[1];
  const int*   src  = edge;
  const int*   dst  = edge + E;
  const float* w_in = (const float*)d_in[2];
  const float* b_in = (const float*)d_in[3];
  const float* ln_g = (const float*)d_in[4];
  const float* ln_b = (const float*)d_in[5];
  const float* W[3]   = {(const float*)d_in[6],  (const float*)d_in[10], (const float*)d_in[14]};
  const float* bL[3]  = {(const float*)d_in[7],  (const float*)d_in[11], (const float*)d_in[15]};
  const float* gL[3]  = {(const float*)d_in[8],  (const float*)d_in[12], (const float*)d_in[16]};
  const float* beL[3] = {(const float*)d_in[9],  (const float*)d_in[13], (const float*)d_in[17]};
  const float* w_head = (const float*)d_in[18];
  const float* b_head = (const float*)d_in[19];
  float* out = (float*)d_out;

  // workspace carve-up (256B aligned)
  char* ws = (char*)d_ws;
  size_t off = 0;
  auto take = [&](size_t bytes) -> char* {
    char* p = ws + off;
    off = (off + bytes + 255) & ~(size_t)255;
    return p;
  };
  float*          dinv  = (float*)take((size_t)N * 4);
  unsigned short* wt[3];
  for (int i = 0; i < 3; ++i) wt[i] = (unsigned short*)take((size_t)HDIM * HDIM * 2);
  unsigned short* xb    = (unsigned short*)take((size_t)N * HDIM * 2);
  float*          xw    = (float*)take((size_t)N * HDIM * 4);
  float*          agg   = (float*)take((size_t)N * HDIM * 4);
  float*          bnsum = (float*)take(2 * HDIM * 4);
  float*          bncf  = (float*)take(2 * HDIM * 4);

  // degrees -> dinv (in place)
  hipMemsetAsync(dinv, 0, (size_t)N * 4, stream);
  deg_count<<<(E + 255) / 256, 256, 0, stream>>>(dst, (unsigned int*)dinv, E);
  deg_to_dinv<<<(N + 255) / 256, 256, 0, stream>>>(dinv, N);

  // pack weights (bf16, transposed)
  for (int i = 0; i < 3; ++i)
    pack_wt<<<(HDIM * HDIM) / 256, 256, 0, stream>>>(W[i], wt[i]);

  // input proj + LN + ELU -> bf16 activations
  input_proj<<<N, HDIM, 0, stream>>>(ctrl, w_in, b_in, ln_g, ln_b, xb);

  for (int i = 0; i < 3; ++i) {
    gemm_xw<<<(N + 15) / 16, 128, 0, stream>>>(xb, wt[i], xw, N);
    init_agg<<<N, HDIM, 0, stream>>>(xw, dinv, bL[i], agg);
    scatter<<<(E + 7) / 8, 256, 0, stream>>>(src, dst, dinv, xw, agg, E);
    hipMemsetAsync(bnsum, 0, 2 * HDIM * 4, stream);
    bn_stats<<<(N + 63) / 64, HDIM, 0, stream>>>(agg, bnsum, N);
    bn_coef<<<1, HDIM, 0, stream>>>(bnsum, gL[i], beL[i], bncf, N);
    bn_apply<<<N, HDIM, 0, stream>>>(agg, bncf, xb);
  }

  head_k<<<N, HDIM, 0, stream>>>(xb, w_head, b_head, out);
}
